// GroupedQueryAttention_36438502539934
// MI455X (gfx1250) — matrix operand order repaired and checked by reading, not compiled, not run on hardware
//
#include <hip/hip_runtime.h>
#include <cstdint>
#include <cstddef>

// ---------------------------------------------------------------------------
// GQA forward for MI455X (gfx1250): all matmuls via v_wmma_f32_16x16x32_bf16.
// B=4, S=1024, HID=2048, NH=32, NKV=8, HD=64. Compute-bound (~120 GFLOP vs
// ~100MB traffic; whole 76MB working set is L2-resident on 192MB L2):
//  - weights pre-packed to bf16 [N][K] once (amortized 32x reuse)
//  - hidden pre-converted to bf16 once (amortized 16x reuse)
//  - GEMM K-loop: double-buffered LDS, register prefetch, global_prefetch_b8,
//    last stage peeled so the hot loop is branch-free
//  - RoPE fused into the Q/K projection epilogue (in-register, pair d/d+32)
//  - flash attention with WMMA for QK^T and PV, online softmax
// ---------------------------------------------------------------------------

typedef __attribute__((ext_vector_type(16))) __bf16       v16bf;
typedef __attribute__((ext_vector_type(8)))  float        v8f;
typedef __attribute__((ext_vector_type(4)))  float        v4f;
typedef __attribute__((ext_vector_type(4)))  unsigned int v4u;

union Frag { v16bf v; v4u q[2]; };

__device__ __forceinline__ unsigned short f2bf(float x) {
    unsigned u = __float_as_uint(x);
    u += 0x7FFFu + ((u >> 16) & 1u);          // round-to-nearest-even
    return (unsigned short)(u >> 16);
}

__device__ __forceinline__ v8f v8f_zero() {
    v8f z = {0.f, 0.f, 0.f, 0.f, 0.f, 0.f, 0.f, 0.f};
    return z;
}

#define WMMA_BF16(A, B, C) \
    __builtin_amdgcn_wmma_f32_16x16x32_bf16(false, (A), false, (B), (short)0, (C), false, false)

// ---------------------------------------------------------------------------
// One-shot: fp32 -> bf16 flat convert (hidden states). 8 elems / thread.
// ---------------------------------------------------------------------------
__global__ __launch_bounds__(256)
void f32_to_bf16(const float* __restrict__ src, unsigned short* __restrict__ dst)
{
    const size_t i = (size_t)blockIdx.x * 256 + threadIdx.x;
    const v4f* s = (const v4f*)src + i * 2;
    v4f g0 = s[0], g1 = s[1];
    v4u p;
    p.x = (unsigned)f2bf(g0[0]) | ((unsigned)f2bf(g0[1]) << 16);
    p.y = (unsigned)f2bf(g0[2]) | ((unsigned)f2bf(g0[3]) << 16);
    p.z = (unsigned)f2bf(g1[0]) | ((unsigned)f2bf(g1[1]) << 16);
    p.w = (unsigned)f2bf(g1[2]) | ((unsigned)f2bf(g1[3]) << 16);
    ((v4u*)dst)[i] = p;
}

// ---------------------------------------------------------------------------
// One-shot: W fp32 [K][N] -> Wt bf16 [N][K]  (32x32 LDS tile transpose).
// Grid: (N/32, K/32), 256 threads.
// ---------------------------------------------------------------------------
__global__ __launch_bounds__(256)
void transpose_w_bf16(const float* __restrict__ W, unsigned short* __restrict__ Wt,
                      int K, int N)
{
    __shared__ float T[32][33];
    const int tx = threadIdx.x & 31, ty = threadIdx.x >> 5;
    const int n0 = blockIdx.x * 32, k0 = blockIdx.y * 32;
#pragma unroll
    for (int i = 0; i < 4; i++)
        T[ty + i * 8][tx] = W[(size_t)(k0 + ty + i * 8) * N + n0 + tx];
    __syncthreads();
#pragma unroll
    for (int i = 0; i < 4; i++)
        Wt[(size_t)(n0 + ty + i * 8) * K + k0 + tx] = f2bf(T[tx][ty + i * 8]);
}

// ---------------------------------------------------------------------------
// Tiled GEMM: C(M,N) = A(M,K) * W(K,N), A bf16 [M][K], Wt bf16 [N][K].
// Block: 256 threads (8 waves), tile 128x128, K-step 32, double-buffered LDS,
// last K-stage peeled (hot loop: unconditional load/compute/store/barrier).
// Wave tile 32x64 (2x4 of 16x16); wave column tile == one head (HD=64) so the
// epilogue applies RoPE fully in-register.
// mode 0: fp32 row-major C              (final projection -> d_out)
// mode 1: RoPE + bf16 scatter [b,h,s,d] (Q and K)
// mode 2: bf16 scatter [b,h,s,d]        (V)
// ---------------------------------------------------------------------------
#define BM  128
#define BN  128
#define BK  32
#define LDT 40   // LDS row stride in halves: 32 + 8 pad (keeps 16B alignment)

__global__ __launch_bounds__(256)
void gemm_bf16_wmma(const unsigned short* __restrict__ A,
                    const unsigned short* __restrict__ Wt,
                    float* __restrict__ C, unsigned short* __restrict__ OutB,
                    int M, int N, int K, int mode, int headsOut)
{
    __shared__ unsigned short At[2][BM * LDT];   // A tile, [row][k]
    __shared__ unsigned short Wl[2][BN * LDT];   // W tile, [n][k]

    const int tid    = threadIdx.x;
    const int lane   = tid & 31;
    const int lane16 = lane & 15;
    const int hh     = (lane >> 4) & 1;
    const int wave   = tid >> 5;
    const int rowBase = (wave & 3) * 32;
    const int colBase = (wave >> 2) * 64;
    const int mBlock  = blockIdx.y * BM;
    const int nBlock  = blockIdx.x * BN;

    v8f acc[2][4];
#pragma unroll
    for (int i = 0; i < 2; i++)
#pragma unroll
        for (int j = 0; j < 4; j++) acc[i][j] = v8f_zero();

    // staging: 128 rows x 2 segments of 16 halves (32B) each, both tiles
    const int ldRow = tid >> 1, seg = tid & 1;
    const unsigned short* aPtr = A  + (size_t)(mBlock + ldRow) * K + seg * 16;
    const unsigned short* wPtr = Wt + (size_t)(nBlock + ldRow) * K + seg * 16;
    const int ldsOff = ldRow * LDT + seg * 16;

    const int nStages = K / BK;

    // prologue: stage 0 -> LDS buffer 0
    {
        v4u a0 = *(const v4u*)aPtr, a1 = *(const v4u*)(aPtr + 8);
        v4u w0 = *(const v4u*)wPtr, w1 = *(const v4u*)(wPtr + 8);
        v4u* da = (v4u*)(At[0] + ldsOff);
        v4u* dw = (v4u*)(Wl[0] + ldsOff);
        da[0] = a0; da[1] = a1;
        dw[0] = w0; dw[1] = w1;
    }
    __syncthreads();

    int cur = 0;
    // ---- main loop: stages 0 .. nStages-2, branch-free body ----
    for (int s = 0; s < nStages - 1; s++) {
        // register prefetch of stage s+1 (unconditional)
        const unsigned short* ap = aPtr + (size_t)(s + 1) * BK;
        const unsigned short* wp = wPtr + (size_t)(s + 1) * BK;
        v4u na0 = *(const v4u*)ap, na1 = *(const v4u*)(ap + 8);
        v4u nw0 = *(const v4u*)wp, nw1 = *(const v4u*)(wp + 8);
        if (s + 2 < nStages) {               // L2 prefetch of stage s+2
            __builtin_prefetch(aPtr + (size_t)(s + 2) * BK, 0, 1);
            __builtin_prefetch(wPtr + (size_t)(s + 2) * BK, 0, 1);
        }

        // compute stage s from LDS[cur]
        Frag af[2];
#pragma unroll
        for (int i = 0; i < 2; i++) {
            const unsigned short* p = At[cur] + (rowBase + i * 16 + lane16) * LDT + hh * 8;
            af[i].q[0] = *(const v4u*)p;
            af[i].q[1] = *(const v4u*)(p + 16);
        }
#pragma unroll
        for (int j = 0; j < 4; j++) {
            const unsigned short* p = Wl[cur] + (colBase + j * 16 + lane16) * LDT + hh * 8;
            Frag bfr;
            bfr.q[0] = *(const v4u*)p;
            bfr.q[1] = *(const v4u*)(p + 16);
#pragma unroll
            for (int i = 0; i < 2; i++)
                acc[i][j] = WMMA_BF16(af[i].v, bfr.v, acc[i][j]);
        }

        // store stage s+1 into the other buffer, flip
        v4u* da = (v4u*)(At[cur ^ 1] + ldsOff);
        v4u* dw = (v4u*)(Wl[cur ^ 1] + ldsOff);
        da[0] = na0; da[1] = na1;
        dw[0] = nw0; dw[1] = nw1;
        __syncthreads();
        cur ^= 1;
    }
    // ---- peeled final stage: compute only ----
    {
        Frag af[2];
#pragma unroll
        for (int i = 0; i < 2; i++) {
            const unsigned short* p = At[cur] + (rowBase + i * 16 + lane16) * LDT + hh * 8;
            af[i].q[0] = *(const v4u*)p;
            af[i].q[1] = *(const v4u*)(p + 16);
        }
#pragma unroll
        for (int j = 0; j < 4; j++) {
            const unsigned short* p = Wl[cur] + (colBase + j * 16 + lane16) * LDT + hh * 8;
            Frag bfr;
            bfr.q[0] = *(const v4u*)p;
            bfr.q[1] = *(const v4u*)(p + 16);
#pragma unroll
            for (int i = 0; i < 2; i++)
                acc[i][j] = WMMA_BF16(af[i].v, bfr.v, acc[i][j]);
        }
    }

    // ------------------------------- epilogue -------------------------------
    if (mode == 0) {
#pragma unroll
        for (int i = 0; i < 2; i++)
#pragma unroll
            for (int j = 0; j < 4; j++)
#pragma unroll
                for (int r = 0; r < 8; r++) {
                    int m = mBlock + rowBase + i * 16 + r + 8 * hh;
                    int n = nBlock + colBase + j * 16 + lane16;
                    C[(size_t)m * N + n] = acc[i][j][r];
                }
    } else if (mode == 1) {
        // RoPE in-register: pair (d, d+32) lives in subtiles (j, j+2) of this wave.
        const int head = (nBlock + colBase) >> 6;
#pragma unroll
        for (int i = 0; i < 2; i++)
#pragma unroll
            for (int r = 0; r < 8; r++) {
                int m    = mBlock + rowBase + i * 16 + r + 8 * hh;
                int bb   = m >> 10;          // / S
                int spos = m & 1023;         // % S
                size_t base = (((size_t)(bb * headsOut + head)) * 1024 + spos) * 64;
#pragma unroll
                for (int j = 0; j < 2; j++) {
                    int   f   = j * 16 + lane16;                 // freq index 0..31
                    float inv = __powf(10000.f, -(float)f * (1.0f / 32.0f));
                    float ang = (float)spos * inv;
                    float sn, cs;
                    __sincosf(ang, &sn, &cs);
                    float a  = acc[i][j][r];
                    float b2 = acc[i][j + 2][r];
                    OutB[base + f]      = f2bf(a * cs - b2 * sn);
                    OutB[base + f + 32] = f2bf(b2 * cs + a * sn);
                }
            }
    } else { // mode 2: plain bf16 scatter [b,h,s,d] (V)
        const int head = (nBlock + colBase) >> 6;
#pragma unroll
        for (int i = 0; i < 2; i++)
#pragma unroll
            for (int r = 0; r < 8; r++) {
                int m    = mBlock + rowBase + i * 16 + r + 8 * hh;
                int bb   = m >> 10;
                int spos = m & 1023;
                size_t base = (((size_t)(bb * headsOut + head)) * 1024 + spos) * 64;
#pragma unroll
                for (int j = 0; j < 4; j++)
                    OutB[base + j * 16 + lane16] = f2bf(acc[i][j][r]);
            }
    }
}

// ---------------------------------------------------------------------------
// Flash attention: grid (S/64, NH, B), 128 threads = 4 waves.
// Wave w owns a 16-row Q tile; block shares K tile + V^T tile in LDS.
// QK^T and PV are both WMMA; online softmax via half-wave shfl_xor reductions
// (rows r / r+8 live in the two 16-lane halves per the C/D VGPR layout).
// Output written bf16 row-major [b*S+s][h*64+d] for the out-projection GEMM.
// ---------------------------------------------------------------------------
#define LDKH 72   // LDS row stride in halves for 64-wide tiles (64 + 8 pad)

__global__ __launch_bounds__(128)
void attn_fwd_wmma(const unsigned short* __restrict__ Qb,
                   const unsigned short* __restrict__ Kb,
                   const unsigned short* __restrict__ Vb,
                   unsigned short* __restrict__ Ob)
{
    __shared__ unsigned short Kt[64 * LDKH];       // [key][dim]
    __shared__ unsigned short Vt[64 * LDKH];       // [dim][key] (transposed)
    __shared__ unsigned short Pw[4 * 16 * LDKH];   // per-wave P tile [row][key]

    const int tid    = threadIdx.x;
    const int lane   = tid & 31;
    const int lane16 = lane & 15;
    const int hh     = (lane >> 4) & 1;
    const int wave   = tid >> 5;

    const int qb64 = blockIdx.x;       // query block of 64 rows
    const int h    = blockIdx.y;       // query head
    const int b    = blockIdx.z;
    const int kvh  = h >> 2;           // N_REP = 4
    const int qrow0 = qb64 * 64 + wave * 16;

    // Q fragments for this wave's 16 rows (resident for whole kernel)
    Frag aq[2];
    {
        const unsigned short* qrow =
            Qb + (((size_t)(b * 32 + h)) * 1024 + qrow0 + lane16) * 64;
#pragma unroll
        for (int c = 0; c < 2; c++) {
            const unsigned short* p = qrow + c * 32 + hh * 8;
            aq[c].q[0] = *(const v4u*)p;
            aq[c].q[1] = *(const v4u*)(p + 16);
        }
    }

    float m_acc[8], l_acc[8];
#pragma unroll
    for (int r = 0; r < 8; r++) { m_acc[r] = -3.0e38f; l_acc[r] = 0.f; }
    v8f Oacc[4];
#pragma unroll
    for (int t = 0; t < 4; t++) Oacc[t] = v8f_zero();

    unsigned short* pw = Pw + wave * 16 * LDKH;
    const int keyld = tid >> 1, segld = tid & 1;   // loaders: 64 keys x 2 halves
    const size_t kvBase = ((size_t)(b * 8 + kvh)) * 1024;

    for (int kb = 0; kb <= qb64; kb++) {
        __syncthreads();   // previous iteration done with Kt/Vt
        // ---- load K tile [key][dim] ----
        {
            const v4u* src = (const v4u*)(Kb + (kvBase + kb * 64 + keyld) * 64 + segld * 32);
            v4u* dst = (v4u*)(Kt + keyld * LDKH + segld * 32);
#pragma unroll
            for (int u = 0; u < 4; u++) dst[u] = src[u];
        }
        // ---- load V tile transposed [dim][key] ----
        {
            const v4u* src = (const v4u*)(Vb + (kvBase + kb * 64 + keyld) * 64 + segld * 32);
            union { v4u q[4]; unsigned short hs[32]; } buf;
#pragma unroll
            for (int u = 0; u < 4; u++) buf.q[u] = src[u];
#pragma unroll
            for (int e = 0; e < 32; e++)
                Vt[(segld * 32 + e) * LDKH + keyld] = buf.hs[e];
        }
        if (kb + 1 <= qb64) {  // L2 prefetch of next K/V block
            __builtin_prefetch(Kb + (kvBase + (kb + 1) * 64 + keyld) * 64 + segld * 32, 0, 1);
            __builtin_prefetch(Vb + (kvBase + (kb + 1) * 64 + keyld) * 64 + segld * 32, 0, 1);
        }
        __syncthreads();

        // ---- S = Q K^T : 4 key-subtiles x 2 k-chunks ----
        v8f st[4];
#pragma unroll
        for (int t = 0; t < 4; t++) {
            st[t] = v8f_zero();
#pragma unroll
            for (int c = 0; c < 2; c++) {
                const unsigned short* p = Kt + (t * 16 + lane16) * LDKH + c * 32 + hh * 8;
                Frag bk;
                bk.q[0] = *(const v4u*)p;
                bk.q[1] = *(const v4u*)(p + 16);
                st[t] = WMMA_BF16(aq[c].v, bk.v, st[t]);
            }
        }

        // ---- scale + causal mask + online softmax ----
        const float SC = 0.125f;                    // HD^-0.5
        const bool diag = (kb == qb64);
        float mb[8];
#pragma unroll
        for (int r = 0; r < 8; r++) {
            float vmax = -3.0e38f;
#pragma unroll
            for (int t = 0; t < 4; t++) {
                float v = st[t][r] * SC;
                if (diag) {
                    int qg = qrow0 + r + 8 * hh;
                    int kg = kb * 64 + t * 16 + lane16;
                    if (kg > qg) v = -3.0e38f;
                }
                st[t][r] = v;
                vmax = fmaxf(vmax, v);
            }
            mb[r] = vmax;
        }
#pragma unroll
        for (int r = 0; r < 8; r++)
#pragma unroll
            for (int off = 8; off >= 1; off >>= 1)   // stays within 16-lane half
                mb[r] = fmaxf(mb[r], __shfl_xor(mb[r], off, 32));

#pragma unroll
        for (int r = 0; r < 8; r++) {
            float mn   = fmaxf(m_acc[r], mb[r]);
            float corr = __expf(m_acc[r] - mn);
            m_acc[r] = mn;
            float rs = 0.f;
#pragma unroll
            for (int t = 0; t < 4; t++) {
                float p = __expf(st[t][r] - mn);
                rs += p;
                pw[(r + 8 * hh) * LDKH + t * 16 + lane16] = f2bf(p);
            }
#pragma unroll
            for (int off = 8; off >= 1; off >>= 1)
                rs += __shfl_xor(rs, off, 32);
            l_acc[r] = l_acc[r] * corr + rs;
#pragma unroll
            for (int t = 0; t < 4; t++)
                Oacc[t][r] *= corr;
        }

        // ---- O += P V : P reloaded from wave-private LDS in A layout ----
        Frag ap[2];
#pragma unroll
        for (int c = 0; c < 2; c++) {
            const unsigned short* p = pw + lane16 * LDKH + c * 32 + hh * 8;
            ap[c].q[0] = *(const v4u*)p;
            ap[c].q[1] = *(const v4u*)(p + 16);
        }
#pragma unroll
        for (int t = 0; t < 4; t++)
#pragma unroll
            for (int c = 0; c < 2; c++) {
                const unsigned short* vp = Vt + (t * 16 + lane16) * LDKH + c * 32 + hh * 8;
                Frag bv;
                bv.q[0] = *(const v4u*)vp;
                bv.q[1] = *(const v4u*)(vp + 16);
                Oacc[t] = WMMA_BF16(ap[c].v, bv.v, Oacc[t]);
            }
    }

    // ---- normalize + store bf16 [b*S + s][h*64 + d] for the out projection ----
#pragma unroll
    for (int t = 0; t < 4; t++)
#pragma unroll
        for (int r = 0; r < 8; r++) {
            int qrow = qrow0 + r + 8 * hh;
            float val = Oacc[t][r] / l_acc[r];
            Ob[((size_t)(b * 1024 + qrow)) * 2048 + h * 64 + t * 16 + lane16] = f2bf(val);
        }
}

// ---------------------------------------------------------------------------
// Host launcher. Inputs (setup_inputs order):
//  0 hidden_states f32 [4,1024,2048]   3 Wq f32 [2048,2048]
//  1 attention_mask (unused, causal)   4 Wk f32 [2048,512]
//  2 position_ids (unused, arange)     5 Wv f32 [2048,512]
//                                      6 Wo f32 [2048,2048]
// Workspace (76 MB):
//  Xb 16MB | WtQ 8MB | WtK 2MB | WtV 2MB | WtO 8MB | Qb 16MB | Kb 4MB |
//  Vb 4MB | Ob 16MB
// ---------------------------------------------------------------------------
extern "C" void kernel_launch(void* const* d_in, const int* in_sizes, int n_in,
                              void* d_out, int out_size, void* d_ws, size_t ws_size,
                              hipStream_t stream)
{
    (void)in_sizes; (void)n_in; (void)out_size; (void)ws_size;

    const float* hidden = (const float*)d_in[0];
    const float* Wq     = (const float*)d_in[3];
    const float* Wk     = (const float*)d_in[4];
    const float* Wv     = (const float*)d_in[5];
    const float* Wo     = (const float*)d_in[6];
    float* out = (float*)d_out;

    char* ws = (char*)d_ws;
    unsigned short* Xb  = (unsigned short*)(ws);               // 16 MB
    unsigned short* WtQ = (unsigned short*)(ws + (16u << 20)); //  8 MB
    unsigned short* WtK = (unsigned short*)(ws + (24u << 20)); //  2 MB
    unsigned short* WtV = (unsigned short*)(ws + (26u << 20)); //  2 MB
    unsigned short* WtO = (unsigned short*)(ws + (28u << 20)); //  8 MB
    unsigned short* Qb  = (unsigned short*)(ws + (36u << 20)); // 16 MB
    unsigned short* Kb  = (unsigned short*)(ws + (52u << 20)); //  4 MB
    unsigned short* Vb  = (unsigned short*)(ws + (56u << 20)); //  4 MB
    unsigned short* Ob  = (unsigned short*)(ws + (60u << 20)); // 16 MB

    const int M = 4096, K = 2048;

    // one-shot packs (amortized over heavy reuse in the GEMMs)
    f32_to_bf16<<<dim3((M * K) / (256 * 8)), 256, 0, stream>>>(hidden, Xb);
    transpose_w_bf16<<<dim3(2048 / 32, K / 32), 256, 0, stream>>>(Wq, WtQ, K, 2048);
    transpose_w_bf16<<<dim3(512 / 32,  K / 32), 256, 0, stream>>>(Wk, WtK, K, 512);
    transpose_w_bf16<<<dim3(512 / 32,  K / 32), 256, 0, stream>>>(Wv, WtV, K, 512);
    transpose_w_bf16<<<dim3(2048 / 32, K / 32), 256, 0, stream>>>(Wo, WtO, K, 2048);

    // Q/K/V projections (RoPE fused into Q/K epilogues)
    gemm_bf16_wmma<<<dim3(2048 / BN, M / BM), 256, 0, stream>>>(
        Xb, WtQ, nullptr, Qb, M, 2048, K, /*mode=*/1, /*heads=*/32);
    gemm_bf16_wmma<<<dim3(512 / BN, M / BM), 256, 0, stream>>>(
        Xb, WtK, nullptr, Kb, M, 512, K, /*mode=*/1, /*heads=*/8);
    gemm_bf16_wmma<<<dim3(512 / BN, M / BM), 256, 0, stream>>>(
        Xb, WtV, nullptr, Vb, M, 512, K, /*mode=*/2, /*heads=*/8);

    // Flash attention (causal applied analytically)
    attn_fwd_wmma<<<dim3(16, 32, 4), 128, 0, stream>>>(Qb, Kb, Vb, Ob);

    // Output projection -> fp32 d_out
    gemm_bf16_wmma<<<dim3(2048 / BN, M / BM), 256, 0, stream>>>(
        Ob, WtO, out, nullptr, M, 2048, K, /*mode=*/0, /*heads=*/0);
}
